// MultiHeadAttention_19284403159514
// MI455X (gfx1250) — matrix operand order repaired and checked by reading, not compiled, not run on hardware
//
#include <hip/hip_runtime.h>
#include <hip/hip_bf16.h>

// ---------------------------------------------------------------------------
// MI455X (gfx1250) fused MHA forward.
// bf16 WMMA (v_wmma_f32_16x16x32_bf16) for all 4 projections + QK^T + PV.
// Attention stages K/V tiles into LDS via the Tensor Data Mover
// (tensor_load_to_lds + s_wait_tensorcnt), double-buffered, wave0-issued.
// ---------------------------------------------------------------------------

#define DIM 1024
#define NH  16
#define DH  64
#define TT  2048
#define BB  2

typedef __attribute__((ext_vector_type(16))) __bf16         v16bf;
typedef __attribute__((ext_vector_type(8)))  __bf16         v8bf;
typedef __attribute__((ext_vector_type(8)))  float          v8f;
typedef __attribute__((ext_vector_type(8)))  unsigned short v8u;
typedef __attribute__((ext_vector_type(16))) unsigned short v16u;
typedef __attribute__((ext_vector_type(4)))  unsigned int   v4u;
typedef __attribute__((ext_vector_type(8)))  int            v8i;
typedef __attribute__((ext_vector_type(4)))  int            v4i;

static __device__ __forceinline__ unsigned short f2bf_bits(float f) {
  unsigned u = __builtin_bit_cast(unsigned, f);
  u += 0x7fffu + ((u >> 16) & 1u);          // round-to-nearest-even
  return (unsigned short)(u >> 16);
}

static __device__ __forceinline__ v8bf load8(const unsigned short* p) {
  return __builtin_bit_cast(v8bf, *reinterpret_cast<const v8u*>(p));
}
static __device__ __forceinline__ v16bf load16(const unsigned short* p) {
  return __builtin_bit_cast(v16bf, *reinterpret_cast<const v16u*>(p));
}
static __device__ __forceinline__ v16bf combine(v8bf lo, v8bf hi) {
  union { v16bf v; v8bf h[2]; } u;
  u.h[0] = lo; u.h[1] = hi;
  return u.v;
}
static __device__ __forceinline__ v8f zero8() {
  v8f z = {0.f, 0.f, 0.f, 0.f, 0.f, 0.f, 0.f, 0.f};
  return z;
}
static __device__ __forceinline__ v8f wmma_bf16(v16bf a, v16bf b, v8f c) {
  return __builtin_amdgcn_wmma_f32_16x16x32_bf16(
      /*neg_a=*/false, a, /*neg_b=*/false, b,
      /*c_mod=*/(short)0, c, /*reuse_a=*/false, /*reuse_b=*/false);
}

// xor-shuffle within 16-lane groups via ds_swizzle_b32 (wave32-safe).
template <int M>
static __device__ __forceinline__ float xor16(float x) {
  return __builtin_bit_cast(
      float, __builtin_amdgcn_ds_swizzle(__builtin_bit_cast(int, x),
                                         (M << 10) | 0x1f));
}
static __device__ __forceinline__ float rowmax16(float x) {
  x = fmaxf(x, xor16<1>(x)); x = fmaxf(x, xor16<2>(x));
  x = fmaxf(x, xor16<4>(x)); x = fmaxf(x, xor16<8>(x));
  return x;
}
static __device__ __forceinline__ float rowsum16(float x) {
  x += xor16<1>(x); x += xor16<2>(x); x += xor16<4>(x); x += xor16<8>(x);
  return x;
}

// ---------------------------------------------------------------------------
// Tensor Data Mover: 2D tile load, bf16 elements (data_size=2B).
// D# per CDNA5 ISA (groups 2/3 zero -> 2D tensor).
// This toolchain uses the 6-arg builtin form:
//   (uint32x4 g0, int32x8 g1, int32x4, int32x4, int32x8, i32 cpol)
// ---------------------------------------------------------------------------
static __device__ __forceinline__ void tdm_load_2d(
    unsigned lds_off, const void* gaddr,
    unsigned tensor_d0, unsigned tensor_d1,
    unsigned long long stride0_elems,
    unsigned tile_d0, unsigned tile_d1) {
  unsigned long long ga = (unsigned long long)(size_t)gaddr;
  v4u g0;
  g0.x = 1u;                                        // count=1, user mode
  g0.y = lds_off;                                   // lds_addr[31:0]
  g0.z = (unsigned)(ga & 0xffffffffu);              // global_addr[31:0]
  g0.w = (unsigned)((ga >> 32) & 0x01ffffffu)       // global_addr[56:32]
         | (2u << 30);                              // type=2 ("image")
  v8i g1;
  g1[0] = (int)(1u << 16);                          // data_size=1 -> 2 bytes
  g1[1] = (int)((tensor_d0 & 0xffffu) << 16);       // tensor_dim0[15:0]
  g1[2] = (int)(((tensor_d0 >> 16) & 0xffffu) |
                ((tensor_d1 & 0xffffu) << 16));     // dim0[31:16] | dim1[15:0]
  g1[3] = (int)(((tensor_d1 >> 16) & 0xffffu) |
                ((tile_d0 & 0xffffu) << 16));       // dim1[31:16] | tile0
  g1[4] = (int)(tile_d1 & 0xffffu);                 // tile1, tile2=0
  g1[5] = (int)(unsigned)(stride0_elems & 0xffffffffu);
  g1[6] = (int)(unsigned)((stride0_elems >> 32) & 0xffffu);
  g1[7] = 0;
  v4i z4 = {0, 0, 0, 0};
  v8i z8 = {0, 0, 0, 0, 0, 0, 0, 0};
  __builtin_amdgcn_tensor_load_to_lds(g0, g1, z4, z4, z8, 0);
}

// ---------------------------------------------------------------------------
// fp32 -> bf16 conversion (4 elements / thread, 8B stores)
// ---------------------------------------------------------------------------
__global__ __launch_bounds__(256) void cvt_bf16_kernel(
    const float* __restrict__ src, unsigned short* __restrict__ dst, int n4) {
  int i = blockIdx.x * blockDim.x + threadIdx.x;
  if (i >= n4) return;
  float4 f = reinterpret_cast<const float4*>(src)[i];
  union { unsigned long long u64; unsigned short s[4]; } o;
  o.s[0] = f2bf_bits(f.x); o.s[1] = f2bf_bits(f.y);
  o.s[2] = f2bf_bits(f.z); o.s[3] = f2bf_bits(f.w);
  reinterpret_cast<unsigned long long*>(dst)[i] = o.u64;
}

// ---------------------------------------------------------------------------
// y[m][n] = sum_c A[m][c] * W[n][c]   (A: [M][1024] bf16, W: [1024][1024] bf16)
// 128-thread block = 4 waves in 2x2; each wave owns a 64x64 output tile.
// mode 0: fp32 store to dstF[m][n]            (final out-projection)
// mode 1: RoPE + scale, bf16 -> [b][h][t][dh] (Q with scale=0.125, K scale=1)
// mode 2: bf16 -> [b][h][dh][t]               (V, transposed for B-fragments)
// ---------------------------------------------------------------------------
__global__ __launch_bounds__(128) void proj_wmma_kernel(
    const unsigned short* __restrict__ A,
    const unsigned short* __restrict__ W,
    unsigned short* __restrict__ dstBf,
    float* __restrict__ dstF,
    int mode, float scale) {
  const int lane = threadIdx.x & 31;
  const int w    = threadIdx.x >> 5;
  const int g    = lane >> 4;     // half-wave
  const int col  = lane & 15;
  const int m0   = blockIdx.y * 128 + (w >> 1) * 64;
  const int n0   = blockIdx.x * 128 + (w & 1) * 64;   // 64-aligned -> one head

  v8f acc[4][4];
#pragma unroll
  for (int i = 0; i < 4; ++i)
#pragma unroll
    for (int j = 0; j < 4; ++j) acc[i][j] = zero8();

  for (int k0 = 0; k0 < DIM; k0 += 32) {
    v16bf a[4];
#pragma unroll
    for (int i = 0; i < 4; ++i) {
      const unsigned short* p =
          A + (size_t)(m0 + i * 16 + col) * DIM + k0 + g * 16;
      a[i] = combine(load8(p), load8(p + 8));   // K 0..15 (+16 if g=1)
    }
#pragma unroll
    for (int j = 0; j < 4; ++j) {
      v16bf b = load16(W + (size_t)(n0 + j * 16 + col) * DIM + k0 + g * 16);
#pragma unroll
      for (int i = 0; i < 4; ++i) acc[i][j] = wmma_bf16(a[i], b, acc[i][j]);
    }
  }

  if (mode == 0) {
#pragma unroll
    for (int i = 0; i < 4; ++i)
#pragma unroll
      for (int j = 0; j < 4; ++j)
#pragma unroll
        for (int r = 0; r < 8; ++r) {
          int m = m0 + i * 16 + g * 8 + r;
          int n = n0 + j * 16 + col;
          dstF[(size_t)m * DIM + n] = acc[i][j][r] * scale;
        }
  } else if (mode == 1) {  // RoPE epilogue
    const int h = n0 >> 6;
#pragma unroll
    for (int i = 0; i < 4; ++i) {
#pragma unroll
      for (int r = 0; r < 8; ++r) {
        int m = m0 + i * 16 + g * 8 + r;
        int b = m >> 11;           // / TT
        int t = m & (TT - 1);
#pragma unroll
        for (int j = 0; j < 4; ++j) {
          int   dh  = j * 16 + col;
          float inv = __expf((float)(dh & 31) * -0.28782313662425575f);
          float sv, cv;
          __sincosf((float)t * inv, &sv, &cv);
          float v1  = acc[i][j][r];
          float v2  = acc[i][j ^ 2][r];          // partner at dh ^ 32
          float rot = (dh < 32) ? -v2 : v2;
          float o   = (v1 * cv + rot * sv) * scale;
          dstBf[(((size_t)(b * NH + h) * TT + t) * DH) + dh] = f2bf_bits(o);
        }
      }
    }
  } else {  // mode 2: V transposed
    const int h = n0 >> 6;
#pragma unroll
    for (int i = 0; i < 4; ++i)
#pragma unroll
      for (int r = 0; r < 8; ++r) {
        int m = m0 + i * 16 + g * 8 + r;
        int b = m >> 11;
        int t = m & (TT - 1);
#pragma unroll
        for (int j = 0; j < 4; ++j) {
          int dh = j * 16 + col;
          dstBf[(((size_t)(b * NH + h) * DH + dh) * TT) + t] =
              f2bf_bits(acc[i][j][r] * scale);
        }
      }
  }
}

// ---------------------------------------------------------------------------
// Flash attention: 256-thread block = 8 waves; each wave owns 16 query rows.
// K/V 64-wide tiles staged into LDS by the TDM (double-buffered, wave0 issues
// 2 descriptors per tile, waits TENSORcnt; barrier publishes to all waves).
// Q/K: [bh][t][64] bf16 ; V: [bh][64][t] bf16 ; CTX out: [b][t][1024] bf16.
// softmax scale already folded into Q.
// ---------------------------------------------------------------------------
struct AttnLds {
  unsigned short Ks[2][64 * 64];   // [t_local][dh]
  unsigned short Vs[2][64 * 64];   // [dh][t_local]
  unsigned short P[8][16 * 64];    // per-wave P relayout tile
};

__global__ __launch_bounds__(256) void attn_wmma_kernel(
    const unsigned short* __restrict__ Q,
    const unsigned short* __restrict__ K,
    const unsigned short* __restrict__ V,
    unsigned short* __restrict__ CTX) {
  __shared__ alignas(16) AttnLds lds;

  const int lane  = threadIdx.x & 31;
  const int w     = threadIdx.x >> 5;
  const int g     = lane >> 4;
  const int col   = lane & 15;
  const int bh    = blockIdx.x;
  const int b     = bh >> 4;
  const int h     = bh & 15;
  const int qbase = blockIdx.y * 128 + w * 16;

  const unsigned short* Qb = Q + (size_t)bh * TT * DH;
  const unsigned short* Kb = K + (size_t)bh * TT * DH;
  const unsigned short* Vb = V + (size_t)bh * DH * TT;

  // Q A-fragments (persist across the whole K loop)
  v16bf qf[2];
#pragma unroll
  for (int c = 0; c < 2; ++c) {
    const unsigned short* p = Qb + (size_t)(qbase + col) * DH + c * 32 + g * 16;
    qf[c] = combine(load8(p), load8(p + 8));
  }

  v8f o[4];
#pragma unroll
  for (int j = 0; j < 4; ++j) o[j] = zero8();
  float mrow[8], lrow[8];
#pragma unroll
  for (int r = 0; r < 8; ++r) { mrow[r] = -1e30f; lrow[r] = 0.f; }

  const int NT = TT / 64;
  auto issue_tile = [&](int it) {
    const int buf = it & 1;
    // K tile: 64 rows (t) x 64 (dh contiguous), row stride 64 elems
    tdm_load_2d((unsigned)(size_t)&lds.Ks[buf][0],
                Kb + (size_t)it * 64 * DH,
                /*tensor_d0=*/DH, /*tensor_d1=*/TT, /*stride0=*/DH,
                /*tile_d0=*/64, /*tile_d1=*/64);
    // V tile: 64 rows (dh) x 64 (t contiguous), row stride T elems
    tdm_load_2d((unsigned)(size_t)&lds.Vs[buf][0],
                Vb + (size_t)it * 64,
                /*tensor_d0=*/TT, /*tensor_d1=*/DH, /*stride0=*/TT,
                /*tile_d0=*/64, /*tile_d1=*/64);
  };

  if (w == 0) issue_tile(0);

  for (int it = 0; it < NT; ++it) {
    const int buf = it & 1;
    if (w == 0) {
      if (it + 1 < NT) {
        issue_tile(it + 1);                       // prefetch next tile
        __builtin_amdgcn_s_wait_tensorcnt((short)2);  // current tile landed
      } else {
        __builtin_amdgcn_s_wait_tensorcnt((short)0);
      }
    }
    __syncthreads();   // publish TDM-written LDS tile to all waves

    // S = Q * K^T  (16 x 64 tile, fp32), K fragments from LDS
    v8f s[4];
#pragma unroll
    for (int j = 0; j < 4; ++j) {
      v8f sa = zero8();
#pragma unroll
      for (int c = 0; c < 2; ++c) {
        v16bf bk = load16(&lds.Ks[buf][(j * 16 + col) * 64 + c * 32 + g * 16]);
        sa = wmma_bf16(qf[c], bk, sa);
      }
      s[j] = sa;
    }

    // online softmax
    float alpha[8];
#pragma unroll
    for (int r = 0; r < 8; ++r) {
      float mx = fmaxf(fmaxf(s[0][r], s[1][r]), fmaxf(s[2][r], s[3][r]));
      mx = rowmax16(mx);
      float mnew = fmaxf(mrow[r], mx);
      alpha[r] = __expf(mrow[r] - mnew);
      mrow[r]  = mnew;
    }
#pragma unroll
    for (int j = 0; j < 4; ++j)
#pragma unroll
      for (int r = 0; r < 8; ++r) s[j][r] = __expf(s[j][r] - mrow[r]);
#pragma unroll
    for (int r = 0; r < 8; ++r) {
      float sm = (s[0][r] + s[1][r]) + (s[2][r] + s[3][r]);
      sm = rowsum16(sm);
      lrow[r] = lrow[r] * alpha[r] + sm;
    }
#pragma unroll
    for (int j = 0; j < 4; ++j)
#pragma unroll
      for (int r = 0; r < 8; ++r) o[j][r] *= alpha[r];

    // C-layout -> A-layout relayout of P through per-wave LDS tile
    // (same-wave DS ops are in-order; no cross-wave sharing of P)
#pragma unroll
    for (int j = 0; j < 4; ++j)
#pragma unroll
      for (int r = 0; r < 8; ++r)
        lds.P[w][(g * 8 + r) * 64 + j * 16 + col] = f2bf_bits(s[j][r]);

    v16bf pf[2];
#pragma unroll
    for (int c = 0; c < 2; ++c) {
      const unsigned short* p = &lds.P[w][col * 64 + c * 32 + g * 16];
      pf[c] = combine(load8(p), load8(p + 8));
    }

    // O += P * V, V fragments from LDS
#pragma unroll
    for (int j = 0; j < 4; ++j) {
#pragma unroll
      for (int c = 0; c < 2; ++c) {
        v16bf bv = load16(&lds.Vs[buf][(j * 16 + col) * 64 + c * 32 + g * 16]);
        o[j] = wmma_bf16(pf[c], bv, o[j]);
      }
    }

    __syncthreads();   // all waves done with buf before it is re-filled
  }

  // finalize + store ctx bf16 [b][t][h*64+dh]
#pragma unroll
  for (int r = 0; r < 8; ++r) {
    float inv = 1.0f / lrow[r];
    int   t   = qbase + g * 8 + r;
#pragma unroll
    for (int j = 0; j < 4; ++j)
      CTX[((size_t)(b * TT + t) * DIM) + h * DH + j * 16 + col] =
          f2bf_bits(o[j][r] * inv);
  }
}

// ---------------------------------------------------------------------------
extern "C" void kernel_launch(void* const* d_in, const int* in_sizes, int n_in,
                              void* d_out, int out_size, void* d_ws,
                              size_t ws_size, hipStream_t stream) {
  (void)in_sizes; (void)n_in; (void)out_size; (void)ws_size;
  const float* x  = (const float*)d_in[0];
  const float* wq = (const float*)d_in[1];
  const float* wk = (const float*)d_in[2];
  const float* wv = (const float*)d_in[3];
  const float* wo = (const float*)d_in[4];
  float*       out = (float*)d_out;

  const size_t M = (size_t)BB * TT;  // 4096 tokens
  unsigned short* ws  = (unsigned short*)d_ws;
  unsigned short* xb  = ws;                 // [M][1024] bf16
  unsigned short* wqb = xb  + M * DIM;      // [1024][1024]
  unsigned short* wkb = wqb + (size_t)DIM * DIM;
  unsigned short* wvb = wkb + (size_t)DIM * DIM;
  unsigned short* wob = wvb + (size_t)DIM * DIM;
  unsigned short* qb  = wob + (size_t)DIM * DIM;  // [bh][t][64]
  unsigned short* kb  = qb  + M * DIM;            // [bh][t][64]
  unsigned short* vb  = kb  + M * DIM;            // [bh][64][t]
  unsigned short* ctx = vb  + M * DIM;            // [b][t][1024]

  // fp32 -> bf16
  {
    int n4 = (int)(M * DIM / 4);
    cvt_bf16_kernel<<<(n4 + 255) / 256, 256, 0, stream>>>(x, xb, n4);
    n4 = DIM * DIM / 4;
    cvt_bf16_kernel<<<(n4 + 255) / 256, 256, 0, stream>>>(wq, wqb, n4);
    cvt_bf16_kernel<<<(n4 + 255) / 256, 256, 0, stream>>>(wk, wkb, n4);
    cvt_bf16_kernel<<<(n4 + 255) / 256, 256, 0, stream>>>(wv, wvb, n4);
    cvt_bf16_kernel<<<(n4 + 255) / 256, 256, 0, stream>>>(wo, wob, n4);
  }

  dim3 gp(DIM / 128, (unsigned)(M / 128));  // (8, 32)
  // Q: RoPE + softmax scale folded in; K: RoPE; V: transposed store
  proj_wmma_kernel<<<gp, 128, 0, stream>>>(xb, wqb, qb, nullptr, 1, 0.125f);
  proj_wmma_kernel<<<gp, 128, 0, stream>>>(xb, wkb, kb, nullptr, 1, 1.0f);
  proj_wmma_kernel<<<gp, 128, 0, stream>>>(xb, wvb, vb, nullptr, 2, 1.0f);

  dim3 ga(BB * NH, TT / 128);  // (32, 16)
  attn_wmma_kernel<<<ga, 256, 0, stream>>>(qb, kb, vb, ctx);

  // out = ctx @ w_out^T, fp32
  proj_wmma_kernel<<<gp, 128, 0, stream>>>(ctx, wob, nullptr, out, 0, 1.0f);
}
